// Predictor_26560077758752
// MI455X (gfx1250) — compile-verified, matrix operand order checked by reading, not run on hardware
//
#include <hip/hip_runtime.h>
#include <hip/hip_bf16.h>
#include <math.h>

// ---------------------------------------------------------------------------
// Problem constants
// ---------------------------------------------------------------------------
#define BATCH 2048
#define SEQ   50
#define DIM   128
#define NT    32000
#define NTILES (NT / 16)          // 2000 column tiles
#define NRB    (BATCH / 16)       // 128 row blocks
#define CHUNKS 16                 // column chunks for the norm pass
#define TILES_PER_CHUNK (NTILES / CHUNKS)   // 125
#define TPB 4                     // column tiles per block in out pass
#define EPS 1e-5f

typedef __attribute__((ext_vector_type(16))) __bf16 v16bf;
typedef __attribute__((ext_vector_type(8)))  __bf16 v8bf;
typedef __attribute__((ext_vector_type(4)))  __bf16 v4bf;
typedef __attribute__((ext_vector_type(8)))  float  v8f;

// ---------------------------------------------------------------------------
// Operand chunk layout (16-bit 16x32 WMMA, wave32):
//   lane&15  -> row (A) / column (B)
//   lane>=16 -> K half offset +8
//   elements 0..7  -> K = kc + h + 0..7
//   elements 8..15 -> K = kc + h + 16..23
// ---------------------------------------------------------------------------
__device__ inline v16bf load_a_chunk(const __bf16* __restrict__ rowp, int kc, int h) {
    v8bf lo = *(const v8bf*)(rowp + kc + h);
    v8bf hi = *(const v8bf*)(rowp + kc + h + 16);
    v16bf r;
#pragma unroll
    for (int j = 0; j < 8; ++j) { r[j] = lo[j]; r[j + 8] = hi[j]; }
    return r;
}

// B tile column n = W row (t0+n).  bf16 source: direct vector load.
__device__ inline v16bf load_b_chunk(const __bf16* __restrict__ W, int t0, int lane, int kc) {
    int n = lane & 15;
    int h = (lane >> 4) * 8;
    const __bf16* rp = W + (size_t)(t0 + n) * DIM + kc + h;
    v8bf lo = *(const v8bf*)rp;
    v8bf hi = *(const v8bf*)(rp + 16);
    v16bf r;
#pragma unroll
    for (int j = 0; j < 8; ++j) { r[j] = lo[j]; r[j + 8] = hi[j]; }
    return r;
}

// f32 source fallback: convert on the fly (used only if ws too small for bf16 W).
__device__ inline v16bf load_b_chunk(const float* __restrict__ W, int t0, int lane, int kc) {
    int n = lane & 15;
    int h = (lane >> 4) * 8;
    const float* rp = W + (size_t)(t0 + n) * DIM + kc + h;
    v8f flo = *(const v8f*)rp;
    v8f fhi = *(const v8f*)(rp + 16);
    v16bf r;
#pragma unroll
    for (int j = 0; j < 8; ++j) {
        r[j]     = (__bf16)flo[j];
        r[j + 8] = (__bf16)fhi[j];
    }
    return r;
}

// ---------------------------------------------------------------------------
// 0) One-time f32 -> bf16 conversion of the four W matrices into ws
// ---------------------------------------------------------------------------
__global__ void convert_w_kernel(const float* __restrict__ W1, const float* __restrict__ W2,
                                 const float* __restrict__ W3, const float* __restrict__ W4,
                                 __bf16* __restrict__ Wq) {
    int m = blockIdx.y;
    const float* W = (m == 0) ? W1 : (m == 1) ? W2 : (m == 2) ? W3 : W4;
    __bf16* dst = Wq + (size_t)m * NT * DIM;
    size_t i = ((size_t)blockIdx.x * blockDim.x + threadIdx.x) * 4;
    if (i < (size_t)NT * DIM) {
        float4 f = *(const float4*)(W + i);
        v4bf o;
        o[0] = (__bf16)f.x; o[1] = (__bf16)f.y; o[2] = (__bf16)f.z; o[3] = (__bf16)f.w;
        *(v4bf*)(dst + i) = o;
    }
}

// ---------------------------------------------------------------------------
// 1) data = mean over S, stored bf16
// ---------------------------------------------------------------------------
__global__ void mean_kernel(const float* __restrict__ enc, __bf16* __restrict__ dataq) {
    int b = blockIdx.x;
    int d = threadIdx.x;
    const float* p = enc + (size_t)b * SEQ * DIM + d;
    float s = 0.0f;
#pragma unroll
    for (int i = 0; i < SEQ; ++i) s += p[(size_t)i * DIM];
    dataq[(size_t)b * DIM + d] = (__bf16)(s * (1.0f / (float)SEQ));
}

// ---------------------------------------------------------------------------
// 2) Partial sum-of-squares of logits per (matrix, row) via WMMA.
//    grid = (CHUNKS, NRB, 4), block = 32 (one wave).
//    Tile loop manually unrolled 2x with named (register-resident) B buffers;
//    next tile's loads issue before current tile's WMMAs.
// ---------------------------------------------------------------------------
template <typename WT>
__global__ void __launch_bounds__(32)
norm_partial_kernel(const __bf16* __restrict__ dataq,
                    const WT* __restrict__ W1, const WT* __restrict__ W2,
                    const WT* __restrict__ W3, const WT* __restrict__ W4,
                    float* __restrict__ partial) {
    int lane = threadIdx.x;
    int ch = blockIdx.x, rb = blockIdx.y, m = blockIdx.z;
    const WT* W = (m == 0) ? W1 : (m == 1) ? W2 : (m == 2) ? W3 : W4;

    int r0 = rb * 16;
    const __bf16* rowp = dataq + (size_t)(r0 + (lane & 15)) * DIM;
    int h = (lane >> 4) * 8;

    v16bf A[4];
#pragma unroll
    for (int c = 0; c < 4; ++c) A[c] = load_a_chunk(rowp, c * 32, h);

    float sq[8];
#pragma unroll
    for (int v = 0; v < 8; ++v) sq[v] = 0.0f;

    int tstart = ch * TILES_PER_CHUNK;
    int tend   = tstart + TILES_PER_CHUNK;

    v16bf B0[4], B1[4];
#pragma unroll
    for (int k = 0; k < 4; ++k) B0[k] = load_b_chunk(W, tstart * 16, lane, k * 32);

    int t = tstart;
    for (; t + 1 < tend; t += 2) {
        // prefetch t+1 into B1, compute t from B0
#pragma unroll
        for (int k = 0; k < 4; ++k) B1[k] = load_b_chunk(W, (t + 1) * 16, lane, k * 32);
        {
            v8f acc = {};
#pragma unroll
            for (int k = 0; k < 4; ++k)
                acc = __builtin_amdgcn_wmma_f32_16x16x32_bf16(
                    false, A[k], false, B0[k], (short)0, acc, false, false);
#pragma unroll
            for (int v = 0; v < 8; ++v) sq[v] += acc[v] * acc[v];
        }
        // prefetch t+2 into B0, compute t+1 from B1
        if (t + 2 < tend) {
#pragma unroll
            for (int k = 0; k < 4; ++k) B0[k] = load_b_chunk(W, (t + 2) * 16, lane, k * 32);
        }
        {
            v8f acc = {};
#pragma unroll
            for (int k = 0; k < 4; ++k)
                acc = __builtin_amdgcn_wmma_f32_16x16x32_bf16(
                    false, A[k], false, B1[k], (short)0, acc, false, false);
#pragma unroll
            for (int v = 0; v < 8; ++v) sq[v] += acc[v] * acc[v];
        }
    }
    if (t < tend) {   // odd tail: B0 already holds tile t
        v8f acc = {};
#pragma unroll
        for (int k = 0; k < 4; ++k)
            acc = __builtin_amdgcn_wmma_f32_16x16x32_bf16(
                false, A[k], false, B0[k], (short)0, acc, false, false);
#pragma unroll
        for (int v = 0; v < 8; ++v) sq[v] += acc[v] * acc[v];
    }

    // Reduce over N (lanes) within each 16-lane half; halves hold different rows.
#pragma unroll
    for (int v = 0; v < 8; ++v) {
        float s = sq[v];
        s += __shfl_xor(s, 1, 32);
        s += __shfl_xor(s, 2, 32);
        s += __shfl_xor(s, 4, 32);
        s += __shfl_xor(s, 8, 32);
        sq[v] = s;
    }
    if ((lane & 15) == 0) {
        int rbase = r0 + (lane >> 4) * 8;   // lane 0 -> rows r0..r0+7, lane 16 -> r0+8..+15
#pragma unroll
        for (int v = 0; v < 8; ++v)
            partial[((size_t)m * BATCH + rbase + v) * CHUNKS + ch] = sq[v];
    }
}

// ---------------------------------------------------------------------------
// 3) inv_norm = 1 / max(sqrt(sum), EPS)   for 4*BATCH rows
// ---------------------------------------------------------------------------
__global__ void norm_finalize_kernel(const float* __restrict__ partial,
                                     float* __restrict__ invn) {
    int i = blockIdx.x * blockDim.x + threadIdx.x;
    if (i >= 4 * BATCH) return;
    float s = 0.0f;
#pragma unroll
    for (int c = 0; c < CHUNKS; ++c) s += partial[(size_t)i * CHUNKS + c];
    invn[i] = 1.0f / fmaxf(sqrtf(s), EPS);
}

// ---------------------------------------------------------------------------
// 4) out = tanh(sum_m coef_m * inv_norm_m[row] * (data . W_m[t]))
//    grid = (NTILES/TPB, NRB), block = 32; each wave does TPB 16x16 tiles.
//    m-loop unrolled with static ping-pong B buffers; during m=3 the next
//    tile's m=0 operand is prefetched.
// ---------------------------------------------------------------------------
template <typename WT>
__global__ void __launch_bounds__(32)
out_kernel(const __bf16* __restrict__ dataq,
           const WT* __restrict__ W1, const WT* __restrict__ W2,
           const WT* __restrict__ W3, const WT* __restrict__ W4,
           const float* __restrict__ ca, const float* __restrict__ cb,
           const float* __restrict__ cc, const float* __restrict__ cd,
           const float* __restrict__ invn, float* __restrict__ out) {
    int lane = threadIdx.x;
    int rb = blockIdx.y;
    int r0 = rb * 16;
    int tbase = blockIdx.x * TPB * 16;

    const __bf16* rowp = dataq + (size_t)(r0 + (lane & 15)) * DIM;
    int h = (lane >> 4) * 8;

    v16bf A[4];
#pragma unroll
    for (int c = 0; c < 4; ++c) A[c] = load_a_chunk(rowp, c * 32, h);

    const WT* Ws[4] = {W1, W2, W3, W4};
    float coef[4] = {*ca, *cb, *cc, *cd};

    int roff = (lane >> 4) * 8;  // C layout: VGPR v -> row M = v + roff
    float scale[4][8];
#pragma unroll
    for (int m = 0; m < 4; ++m)
#pragma unroll
        for (int v = 0; v < 8; ++v)
            scale[m][v] = coef[m] * invn[(size_t)m * BATCH + r0 + roff + v];

    int n = lane & 15;

    v16bf B0[4], B1[4];
#pragma unroll
    for (int k = 0; k < 4; ++k) B0[k] = load_b_chunk(Ws[0], tbase, lane, k * 32);

    for (int ti = 0; ti < TPB; ++ti) {
        int t0 = tbase + ti * 16;

        float accv[8];
#pragma unroll
        for (int v = 0; v < 8; ++v) accv[v] = 0.0f;

        // m = 0: B0 -> compute, prefetch m=1 into B1
#pragma unroll
        for (int k = 0; k < 4; ++k) B1[k] = load_b_chunk(Ws[1], t0, lane, k * 32);
        {
            v8f c = {};
#pragma unroll
            for (int k = 0; k < 4; ++k)
                c = __builtin_amdgcn_wmma_f32_16x16x32_bf16(
                    false, A[k], false, B0[k], (short)0, c, false, false);
#pragma unroll
            for (int v = 0; v < 8; ++v) accv[v] += scale[0][v] * c[v];
        }
        // m = 1: B1 -> compute, prefetch m=2 into B0
#pragma unroll
        for (int k = 0; k < 4; ++k) B0[k] = load_b_chunk(Ws[2], t0, lane, k * 32);
        {
            v8f c = {};
#pragma unroll
            for (int k = 0; k < 4; ++k)
                c = __builtin_amdgcn_wmma_f32_16x16x32_bf16(
                    false, A[k], false, B1[k], (short)0, c, false, false);
#pragma unroll
            for (int v = 0; v < 8; ++v) accv[v] += scale[1][v] * c[v];
        }
        // m = 2: B0 -> compute, prefetch m=3 into B1
#pragma unroll
        for (int k = 0; k < 4; ++k) B1[k] = load_b_chunk(Ws[3], t0, lane, k * 32);
        {
            v8f c = {};
#pragma unroll
            for (int k = 0; k < 4; ++k)
                c = __builtin_amdgcn_wmma_f32_16x16x32_bf16(
                    false, A[k], false, B0[k], (short)0, c, false, false);
#pragma unroll
            for (int v = 0; v < 8; ++v) accv[v] += scale[2][v] * c[v];
        }
        // m = 3: B1 -> compute, prefetch next tile's m=0 into B0
        if (ti + 1 < TPB) {
#pragma unroll
            for (int k = 0; k < 4; ++k)
                B0[k] = load_b_chunk(Ws[0], t0 + 16, lane, k * 32);
        }
        {
            v8f c = {};
#pragma unroll
            for (int k = 0; k < 4; ++k)
                c = __builtin_amdgcn_wmma_f32_16x16x32_bf16(
                    false, A[k], false, B1[k], (short)0, c, false, false);
#pragma unroll
            for (int v = 0; v < 8; ++v) accv[v] += scale[3][v] * c[v];
        }

        // epilogue: tanh + store
#pragma unroll
        for (int v = 0; v < 8; ++v) {
            int row = r0 + roff + v;
            out[(size_t)row * NT + t0 + n] = tanhf(accv[v]);
        }
    }
}

// ---------------------------------------------------------------------------
// 5) target: ones, then zeros scattered at event_type-1
// ---------------------------------------------------------------------------
__global__ void fill_ones_kernel(float4* __restrict__ tgt, int n4) {
    int i = blockIdx.x * blockDim.x + threadIdx.x;
    if (i < n4) tgt[i] = make_float4(1.0f, 1.0f, 1.0f, 1.0f);
}

__global__ void scatter_zero_kernel(const int* __restrict__ et, float* __restrict__ tgt) {
    int i = blockIdx.x * blockDim.x + threadIdx.x;
    if (i >= BATCH * SEQ) return;
    int b = i / SEQ;
    int e = et[i];
    if (e > 0 && e <= NT) tgt[(size_t)b * NT + (e - 1)] = 0.0f;
}

// ---------------------------------------------------------------------------
// Launch
// ---------------------------------------------------------------------------
extern "C" void kernel_launch(void* const* d_in, const int* in_sizes, int n_in,
                              void* d_out, int out_size, void* d_ws, size_t ws_size,
                              hipStream_t stream) {
    const float* enc = (const float*)d_in[0];
    const int*   et  = (const int*)d_in[1];
    // d_in[2] = user_type (unused by the reference output)
    const float* W1 = (const float*)d_in[3];
    const float* W2 = (const float*)d_in[4];
    const float* W3 = (const float*)d_in[5];
    const float* W4 = (const float*)d_in[6];
    const float* ca = (const float*)d_in[7];
    const float* cb = (const float*)d_in[8];
    const float* cc = (const float*)d_in[9];
    const float* cd = (const float*)d_in[10];

    float* out = (float*)d_out;                       // (BATCH, NT)
    float* tgt = out + (size_t)BATCH * NT;            // (BATCH, NT)

    // workspace layout
    char* ws = (char*)d_ws;
    size_t off = 0;
    __bf16* dataq = (__bf16*)(ws + off);  off += (size_t)BATCH * DIM * sizeof(__bf16);       // 512 KB
    float* partial = (float*)(ws + off);  off += (size_t)4 * BATCH * CHUNKS * sizeof(float); // 512 KB
    float* invn    = (float*)(ws + off);  off += (size_t)4 * BATCH * sizeof(float);          // 32 KB
    __bf16* Wq     = (__bf16*)(ws + off);
    size_t need_bf16w = off + (size_t)4 * NT * DIM * sizeof(__bf16);                         // ~33.8 MB
    bool use_wq = (ws_size >= need_bf16w);   // fixed per run -> deterministic

    // 1) mean over S
    mean_kernel<<<dim3(BATCH), dim3(DIM), 0, stream>>>(enc, dataq);

    if (use_wq) {
        // 0) one-time f32 -> bf16 conversion of W (L2-resident thereafter)
        convert_w_kernel<<<dim3((NT * DIM / 4 + 255) / 256, 4), dim3(256), 0, stream>>>(
            W1, W2, W3, W4, Wq);
        const __bf16* Q1 = Wq;
        const __bf16* Q2 = Wq + (size_t)1 * NT * DIM;
        const __bf16* Q3 = Wq + (size_t)2 * NT * DIM;
        const __bf16* Q4 = Wq + (size_t)3 * NT * DIM;

        norm_partial_kernel<__bf16><<<dim3(CHUNKS, NRB, 4), dim3(32), 0, stream>>>(
            dataq, Q1, Q2, Q3, Q4, partial);
        norm_finalize_kernel<<<dim3((4 * BATCH + 255) / 256), dim3(256), 0, stream>>>(
            partial, invn);
        out_kernel<__bf16><<<dim3(NTILES / TPB, NRB), dim3(32), 0, stream>>>(
            dataq, Q1, Q2, Q3, Q4, ca, cb, cc, cd, invn, out);
    } else {
        norm_partial_kernel<float><<<dim3(CHUNKS, NRB, 4), dim3(32), 0, stream>>>(
            dataq, W1, W2, W3, W4, partial);
        norm_finalize_kernel<<<dim3((4 * BATCH + 255) / 256), dim3(256), 0, stream>>>(
            partial, invn);
        out_kernel<float><<<dim3(NTILES / TPB, NRB), dim3(32), 0, stream>>>(
            dataq, W1, W2, W3, W4, ca, cb, cc, cd, invn, out);
    }

    // 5) target = ones with scattered zeros
    int n4 = BATCH * NT / 4;
    fill_ones_kernel<<<dim3((n4 + 255) / 256), dim3(256), 0, stream>>>((float4*)tgt, n4);
    scatter_zero_kernel<<<dim3((BATCH * SEQ + 255) / 256), dim3(256), 0, stream>>>(et, tgt);
}
